// DPCA3D_38929583571419
// MI455X (gfx1250) — matrix operand order repaired and checked by reading, not compiled, not run on hardware
//
#include <hip/hip_runtime.h>

typedef __attribute__((ext_vector_type(16))) _Float16 v16h;
typedef __attribute__((ext_vector_type(8)))  _Float16 v8h;
typedef __attribute__((ext_vector_type(2)))  _Float16 h2;
typedef __attribute__((ext_vector_type(8)))  float    v8f;

#define HEADS 8
#define DH    64
#define INNER 512
#define DIMC  256
#define TOPK  512
#define LK    4096
#define LQ    16384
#define NB    2

// ---------------------------------------------------------------------------
// WMMA helpers (CDNA5 wave32, 16x16x32 f16 -> f32)
// ---------------------------------------------------------------------------
__device__ __forceinline__ v8f wmma32(v16h a, v16h b, v8f c) {
  return __builtin_amdgcn_wmma_f32_16x16x32_f16(false, a, false, b, (short)0, c,
                                                false, false);
}

// A fragment: 16(M) x 32(K) tile from a row-major [rows x ld] f16 matrix.
__device__ __forceinline__ v16h load_a(const _Float16* base, int row, int col0,
                                       int ld, int lane) {
  const int ch = (lane & 16) ? 8 : 0;
  const _Float16* p = base + (size_t)row * ld + col0 + ch;
  v8h lo = *(const v8h*)(p);
  v8h hi = *(const v8h*)(p + 16);
  v16h f;
#pragma unroll
  for (int i = 0; i < 8; ++i) { f[i] = lo[i]; f[i + 8] = hi[i]; }
  return f;
}

// B fragment: 32(K) x 16(N); column n of B = row n of the given row-major
// [N x ld] matrix. Works for global or LDS pointers.
__device__ __forceinline__ v16h load_b(const _Float16* base, int nrow, int col0,
                                       int ld, int lane) {
  const int ko = (lane & 16) ? 8 : 0;
  const _Float16* p = base + (size_t)nrow * ld + col0 + ko;
  v8h lo = *(const v8h*)(p);
  v8h hi = *(const v8h*)(p + 16);
  v16h f;
#pragma unroll
  for (int i = 0; i < 8; ++i) { f[i] = lo[i]; f[i + 8] = hi[i]; }
  return f;
}

__device__ __forceinline__ h2 habs2(h2 x) {
  unsigned u = __builtin_bit_cast(unsigned, x) & 0x7fff7fffu;
  return __builtin_bit_cast(h2, u);
}

// ---------------------------------------------------------------------------
// 0. fp32 -> fp16 weight conversion
// ---------------------------------------------------------------------------
__global__ void f32_to_f16(const float* __restrict__ in, _Float16* __restrict__ out,
                           int n) {
  int i = blockIdx.x * 256 + threadIdx.x;
  if (i < n) out[i] = (_Float16)in[i];
}

// ---------------------------------------------------------------------------
// 1. channel LayerNorm + transpose: x [b, 256, L] f32 -> xT [b, L, 256] f16
//    LDS-tiled: coalesced loads over positions, coalesced row writes.
// ---------------------------------------------------------------------------
__global__ void ln_transpose(const float* __restrict__ x, const float* __restrict__ g,
                             const float* __restrict__ bt, _Float16* __restrict__ outT,
                             int L) {
  __shared__ float tile[256 * 33];
  __shared__ float red[2][8][32];
  __shared__ float mu_s[32], inv_s[32];
  int t = threadIdx.x, lane = t & 31, wave = t >> 5;
  int pos0 = blockIdx.x * 32;
  int b = blockIdx.y;
  const float* xb = x + (size_t)b * DIMC * L + pos0;
#pragma unroll 4
  for (int i = 0; i < 32; ++i) {
    int c = i * 8 + wave;
    tile[c * 33 + lane] = xb[(size_t)c * L + lane];
  }
  __syncthreads();
  {
    float s = 0.f, s2 = 0.f;
#pragma unroll 8
    for (int i = 0; i < 32; ++i) {
      float v = tile[(wave * 32 + i) * 33 + lane];
      s += v; s2 += v * v;
    }
    red[0][wave][lane] = s; red[1][wave][lane] = s2;
  }
  __syncthreads();
  if (t < 32) {
    float ss = 0.f, ss2 = 0.f;
#pragma unroll
    for (int k = 0; k < 8; ++k) { ss += red[0][k][t]; ss2 += red[1][k][t]; }
    float mu = ss * (1.f / 256.f);
    mu_s[t] = mu;
    inv_s[t] = rsqrtf(ss2 * (1.f / 256.f) - mu * mu + 1e-5f);
  }
  __syncthreads();
  for (int pp = wave; pp < 32; pp += 8) {
    float mu = mu_s[pp], inv = inv_s[pp];
    _Float16* o = outT + ((size_t)b * L + pos0 + pp) * DIMC;
#pragma unroll
    for (int i = 0; i < 8; ++i) {
      int c = i * 32 + lane;
      o[c] = (_Float16)((tile[c * 33 + pp] - mu) * inv * g[c] + bt[c]);
    }
  }
}

// ---------------------------------------------------------------------------
// 2a. fused projection GEMM -> f16 with optional per-head l2norm.
//     C[b][M][512] (f16) = A[b][M][K] (f16) * Bw[512][K]^T; a wave owns a
//     16x64 strip (one head chunk) so the norm is a 16-lane shuffle reduce.
// ---------------------------------------------------------------------------
__global__ void gemm_headfuse(const _Float16* __restrict__ A, size_t strideA,
                              const _Float16* __restrict__ Bw,
                              _Float16* __restrict__ Cf16, size_t strideC,
                              int K, int norm) {
  __shared__ _Float16 ost[8][16 * 64];
  int lane = threadIdx.x & 31, wave = threadIdx.x >> 5;
  int wm = wave >> 2, wn = wave & 3;
  const _Float16* Ab = A + (size_t)blockIdx.z * strideA;
  _Float16* Cb = Cf16 + (size_t)blockIdx.z * strideC;
  int m0 = blockIdx.x * 32 + wm * 16;
  int col0 = (blockIdx.y * 4 + wn) * 64;
  int arow = m0 + (lane & 15);
  v8f acc[4] = {};
  for (int k0 = 0; k0 < K; k0 += 32) {
    v16h a = load_a(Ab, arow, k0, K, lane);
#pragma unroll
    for (int tn = 0; tn < 4; ++tn) {
      v16h b = load_b(Bw, col0 + tn * 16 + (lane & 15), k0, K, lane);
      acc[tn] = wmma32(a, b, acc[tn]);
    }
  }
  int n = lane & 15, mo = (lane & 16) ? 8 : 0;
  float inv[8];
#pragma unroll
  for (int r = 0; r < 8; ++r) {
    if (norm) {
      float s = acc[0][r] * acc[0][r] + acc[1][r] * acc[1][r] +
                acc[2][r] * acc[2][r] + acc[3][r] * acc[3][r];
#pragma unroll
      for (int msk = 1; msk < 16; msk <<= 1) s += __shfl_xor(s, msk, 32);
      inv[r] = 1.f / fmaxf(sqrtf(s), 1e-12f);
    } else {
      inv[r] = 1.f;
    }
  }
  _Float16* os = ost[wave];
#pragma unroll
  for (int r = 0; r < 8; ++r)
#pragma unroll
    for (int tn = 0; tn < 4; ++tn)
      os[(r + mo) * 64 + tn * 16 + n] = (_Float16)(acc[tn][r] * inv[r]);
  // same-wave DS ops are in-order: coalesced b32 row stores
  const unsigned* osw = (const unsigned*)os;
  unsigned* obw = (unsigned*)(Cb + (size_t)m0 * INNER + col0);
#pragma unroll
  for (int r = 0; r < 16; ++r)
    obw[(size_t)r * (INNER / 2) + lane] = osw[r * 32 + lane];
}

// ---------------------------------------------------------------------------
// 2b. generic GEMM -> f32 (used for the output projection).
// ---------------------------------------------------------------------------
__global__ void gemm_abT(const _Float16* __restrict__ A, size_t strideA,
                         const _Float16* __restrict__ Bw,
                         float* __restrict__ C, size_t strideC, int N, int K) {
  int lane = threadIdx.x & 31, wave = threadIdx.x >> 5;
  int wm = wave >> 2, wn = wave & 3;
  const _Float16* Ab = A + (size_t)blockIdx.z * strideA;
  float* Cb = C + (size_t)blockIdx.z * strideC;
  int m0 = blockIdx.x * 32 + wm * 16;
  int n0 = blockIdx.y * 64 + wn * 16;
  int arow = m0 + (lane & 15);
  int brow = n0 + (lane & 15);
  v8f acc = {};
  for (int k0 = 0; k0 < K; k0 += 32) {
    v16h a = load_a(Ab, arow, k0, K, lane);
    v16h b = load_b(Bw, brow, k0, K, lane);
    acc = wmma32(a, b, acc);
  }
  int n = lane & 15, mo = (lane & 16) ? 8 : 0;
#pragma unroll
  for (int r = 0; r < 8; ++r)
    Cb[(size_t)(m0 + mo + r) * N + n0 + n] = acc[r];
}

// ---------------------------------------------------------------------------
// 3. L1 cdist to 512 anchors, min over anchors. Packed-f16 VALU math:
//    per 2 elems: v_pk_add(neg) + v_and(abs) + v_pk_add(acc).
// ---------------------------------------------------------------------------
__global__ void l1_min(const _Float16* __restrict__ qf16,
                       const _Float16* __restrict__ kf16,
                       const int* __restrict__ rand_idx,
                       float* __restrict__ min_d) {
  __shared__ _Float16 Qs[TOPK * DH];  // 64 KB
  int bh = blockIdx.y;
  int b = bh >> 3, h = bh & 7;
  for (int t = threadIdx.x; t < TOPK; t += 256) {
    int qi = rand_idx[bh * TOPK + t];
    const _Float16* qp = qf16 + ((size_t)b * LQ + qi) * INNER + h * DH;
    for (int d = 0; d < DH; ++d) Qs[t * DH + d] = qp[d];
  }
  __syncthreads();
  int key = blockIdx.x * 256 + threadIdx.x;
  const h2* kp = (const h2*)(kf16 + ((size_t)b * LK + key) * INNER + h * DH);
  h2 kv[DH / 2];
#pragma unroll
  for (int d = 0; d < DH / 2; ++d) kv[d] = kp[d];
  float best = 1e30f;
  for (int t = 0; t < TOPK; ++t) {
    const h2* qp = (const h2*)(Qs + t * DH);
    h2 a0 = {}, a1 = {}, a2 = {}, a3 = {};
#pragma unroll
    for (int i = 0; i < DH / 2; i += 4) {
      a0 += habs2(kv[i + 0] - qp[i + 0]);
      a1 += habs2(kv[i + 1] - qp[i + 1]);
      a2 += habs2(kv[i + 2] - qp[i + 2]);
      a3 += habs2(kv[i + 3] - qp[i + 3]);
    }
    h2 a = (a0 + a1) + (a2 + a3);
    float s = (float)a[0] + (float)a[1];
    best = fminf(best, s);
  }
  min_d[(size_t)bh * LK + key] = best;
}

// ---------------------------------------------------------------------------
// 4. exact 512-smallest-of-4096 selection (bit-descent + ordered ties).
// ---------------------------------------------------------------------------
__global__ void select_topk(const float* __restrict__ min_d, int* __restrict__ idx) {
  __shared__ unsigned sh[LK];
  __shared__ int cnt;
  __shared__ int sa[256], sb[256];
  int bh = blockIdx.x;
  int t = threadIdx.x;
  for (int i = t; i < LK; i += 256) sh[i] = __float_as_uint(min_d[(size_t)bh * LK + i]);
  __syncthreads();
  unsigned T = 0;
  for (int bit = 31; bit >= 0; --bit) {
    unsigned cand = T | (1u << bit);
    if (t == 0) cnt = 0;
    __syncthreads();
    int c = 0;
    for (int i = t; i < LK; i += 256) c += (sh[i] < cand) ? 1 : 0;
    atomicAdd(&cnt, c);
    __syncthreads();
    if (cnt < TOPK) T = cand;
    __syncthreads();
  }
  const int chunk = LK / 256;
  int nl = 0, ne = 0;
  for (int i = t * chunk; i < (t + 1) * chunk; ++i) {
    nl += (sh[i] < T) ? 1 : 0;
    ne += (sh[i] == T) ? 1 : 0;
  }
  sa[t] = nl; sb[t] = ne;
  __syncthreads();
  for (int off = 1; off < 256; off <<= 1) {
    int av = (t >= off) ? sa[t - off] : 0;
    int bv = (t >= off) ? sb[t - off] : 0;
    __syncthreads();
    sa[t] += av; sb[t] += bv;
    __syncthreads();
  }
  int pless = sa[t] - nl;
  int peq = sb[t] - ne;
  int total_less = sa[255];
  int need = TOPK - total_less;
  int* out = idx + bh * TOPK;
  int li = pless, ei = peq;
  for (int i = t * chunk; i < (t + 1) * chunk; ++i) {
    unsigned x = sh[i];
    if (x < T) {
      out[li++] = i;
    } else if (x == T) {
      if (ei < need) out[total_less + ei] = i;
      ++ei;
    }
  }
}

// ---------------------------------------------------------------------------
// 5. gather selected keys/values: Ksel [bh][512][64], Vt [bh][64][512].
// ---------------------------------------------------------------------------
__global__ void gather_kv(const _Float16* __restrict__ kf16,
                          const _Float16* __restrict__ vf16,
                          const int* __restrict__ idx,
                          _Float16* __restrict__ Ksel, _Float16* __restrict__ Vt) {
  int bh = blockIdx.y;
  int b = bh >> 3, h = bh & 7;
  int j = blockIdx.x * 256 + threadIdx.x;
  int row = idx[bh * TOPK + j];
  const _Float16* kp = kf16 + ((size_t)b * LK + row) * INNER + h * DH;
  _Float16* ko = Ksel + ((size_t)bh * TOPK + j) * DH;
#pragma unroll
  for (int d = 0; d < DH; ++d) ko[d] = kp[d];
  const _Float16* vp = vf16 + ((size_t)b * LK + row) * INNER + h * DH;
  _Float16* vo = Vt + (size_t)bh * DH * TOPK;
#pragma unroll
  for (int d = 0; d < DH; ++d) vo[(size_t)d * TOPK + j] = vp[d];
}

// ---------------------------------------------------------------------------
// 6. flash attention. K/V staged in LDS per 128-key chunk (shared by all 8
//    waves); probs bounce through a wave-private LDS slab to become an A
//    fragment; output staged in LDS and stored as coalesced b32 rows.
// ---------------------------------------------------------------------------
__global__ void attn_kernel(const _Float16* __restrict__ qf16,
                            const _Float16* __restrict__ Ksel,
                            const _Float16* __restrict__ Vt,
                            _Float16* __restrict__ outT) {
  __shared__ _Float16 Kst[128 * 64];       // 16 KB: 128 keys x 64 dims
  __shared__ _Float16 Vst[64 * 128];       // 16 KB: 64 dims x 128 keys
  __shared__ _Float16 stage[8][16 * 32];   //  8 KB: probs staging
  __shared__ _Float16 ostage[8][16 * 64];  // 16 KB: output staging
  int t = threadIdx.x, lane = t & 31, wave = t >> 5;
  int bh = blockIdx.y;
  int b = bh >> 3, h = bh & 7;
  int qrow0 = blockIdx.x * 128 + wave * 16;
  _Float16* st = stage[wave];

  const _Float16* Ab = qf16 + (size_t)b * LQ * INNER;
  const _Float16* Kb = Ksel + (size_t)bh * TOPK * DH;
  const _Float16* Vb = Vt + (size_t)bh * DH * TOPK;

  int arow = qrow0 + (lane & 15);
  v16h aq0 = load_a(Ab, arow, h * DH, INNER, lane);
  v16h aq1 = load_a(Ab, arow, h * DH + 32, INNER, lane);

  v8f o0 = {}, o1 = {}, o2 = {}, o3 = {};
  float m[8], l[8];
#pragma unroll
  for (int r = 0; r < 8; ++r) { m[r] = -1e30f; l[r] = 0.f; }

  int brow = lane & 15;
  int n = lane & 15, mo = (lane & 16) ? 8 : 0;
  int ch = (lane & 16) ? 8 : 0;

  for (int cc = 0; cc < 4; ++cc) {  // 4 chunks of 128 keys
    __syncthreads();
    {
      const uint4* ks = (const uint4*)(Kb + (size_t)cc * 128 * DH);
      uint4* kd = (uint4*)Kst;
      for (int i = t; i < 1024; i += 256) kd[i] = ks[i];
      uint4* vd = (uint4*)Vst;
      for (int i = t; i < 1024; i += 256) {
        int row = i >> 4, seg = i & 15;
        vd[i] = *(const uint4*)(Vb + (size_t)row * TOPK + cc * 128 + seg * 8);
      }
    }
    __syncthreads();
#pragma unroll
    for (int j2 = 0; j2 < 4; ++j2) {  // 4 pairs of 16-key tiles per chunk
      int kloc = j2 * 32;
      v8f s0 = {}, s1 = {};
      v16h bk;
      bk = load_b(Kst, kloc + brow, 0, DH, lane);       s0 = wmma32(aq0, bk, s0);
      bk = load_b(Kst, kloc + brow, 32, DH, lane);      s0 = wmma32(aq1, bk, s0);
      bk = load_b(Kst, kloc + 16 + brow, 0, DH, lane);  s1 = wmma32(aq0, bk, s1);
      bk = load_b(Kst, kloc + 16 + brow, 32, DH, lane); s1 = wmma32(aq1, bk, s1);

#pragma unroll
      for (int r = 0; r < 8; ++r) {
        float tm = fmaxf(s0[r], s1[r]);
#pragma unroll
        for (int msk = 1; msk < 16; msk <<= 1) tm = fmaxf(tm, __shfl_xor(tm, msk, 32));
        float nm = fmaxf(m[r], tm);
        float alpha = __expf(m[r] - nm);
        o0[r] *= alpha; o1[r] *= alpha; o2[r] *= alpha; o3[r] *= alpha;
        float p0 = __expf(s0[r] - nm);
        float p1 = __expf(s1[r] - nm);
        float ts = p0 + p1;
#pragma unroll
        for (int msk = 1; msk < 16; msk <<= 1) ts += __shfl_xor(ts, msk, 32);
        l[r] = l[r] * alpha + ts;
        m[r] = nm;
        st[(r + mo) * 32 + n] = (_Float16)p0;
        st[(r + mo) * 32 + 16 + n] = (_Float16)p1;
      }
      v16h ap;  // probs as A fragment (same-wave DS ops are in-order)
      {
        const _Float16* p = st + (lane & 15) * 32 + ch;
#pragma unroll
        for (int i = 0; i < 8; ++i) { ap[i] = p[i]; ap[i + 8] = p[i + 16]; }
      }
      v16h bv;
      bv = load_b(Vst, 0 * 16 + brow, kloc, 128, lane); o0 = wmma32(ap, bv, o0);
      bv = load_b(Vst, 1 * 16 + brow, kloc, 128, lane); o1 = wmma32(ap, bv, o1);
      bv = load_b(Vst, 2 * 16 + brow, kloc, 128, lane); o2 = wmma32(ap, bv, o2);
      bv = load_b(Vst, 3 * 16 + brow, kloc, 128, lane); o3 = wmma32(ap, bv, o3);
    }
  }

  _Float16* os = ostage[wave];
#pragma unroll
  for (int r = 0; r < 8; ++r) {
    float inv = 1.f / l[r];
    os[(r + mo) * 64 + 0 + n]  = (_Float16)(o0[r] * inv);
    os[(r + mo) * 64 + 16 + n] = (_Float16)(o1[r] * inv);
    os[(r + mo) * 64 + 32 + n] = (_Float16)(o2[r] * inv);
    os[(r + mo) * 64 + 48 + n] = (_Float16)(o3[r] * inv);
  }
  const unsigned* osw = (const unsigned*)os;
  unsigned* obw = (unsigned*)(outT + ((size_t)b * LQ + qrow0) * INNER + h * DH);
#pragma unroll
  for (int r = 0; r < 16; ++r)
    obw[(size_t)r * (INNER / 2) + lane] = osw[r * 32 + lane];
}

// ---------------------------------------------------------------------------
// 7. final channel LN + gamma*x + residual, LDS-tiled transpose for fully
//    coalesced reads and writes.
// ---------------------------------------------------------------------------
__global__ void final_ln(const float* __restrict__ resT, const float* __restrict__ g,
                         const float* __restrict__ bt, const float* __restrict__ gamma,
                         const float* __restrict__ qsrc, float* __restrict__ out,
                         int L) {
  __shared__ float tile[256 * 33];
  __shared__ float red[2][8][32];
  __shared__ float mu_s[32], inv_s[32];
  int t = threadIdx.x, lane = t & 31, wave = t >> 5;
  int pos0 = blockIdx.x * 32;
  int b = blockIdx.y;
  for (int pp = wave; pp < 32; pp += 8) {
    const float* r = resT + ((size_t)b * L + pos0 + pp) * DIMC;
#pragma unroll
    for (int i = 0; i < 8; ++i) {
      int c = i * 32 + lane;
      tile[c * 33 + pp] = r[c];
    }
  }
  __syncthreads();
  {
    float s = 0.f, s2 = 0.f;
#pragma unroll 8
    for (int i = 0; i < 32; ++i) {
      float v = tile[(wave * 32 + i) * 33 + lane];
      s += v; s2 += v * v;
    }
    red[0][wave][lane] = s; red[1][wave][lane] = s2;
  }
  __syncthreads();
  if (t < 32) {
    float ss = 0.f, ss2 = 0.f;
#pragma unroll
    for (int k = 0; k < 8; ++k) { ss += red[0][k][t]; ss2 += red[1][k][t]; }
    float mu = ss * (1.f / 256.f);
    mu_s[t] = mu;
    inv_s[t] = rsqrtf(ss2 * (1.f / 256.f) - mu * mu + 1e-5f);
  }
  __syncthreads();
  float gm = gamma[0];
  const float* qb = qsrc + (size_t)b * DIMC * L + pos0;
  float* ob = out + (size_t)b * DIMC * L + pos0;
#pragma unroll 4
  for (int i = 0; i < 32; ++i) {
    int c = i * 8 + wave;
    float v = (tile[c * 33 + lane] - mu_s[lane]) * inv_s[lane] * g[c] + bt[c];
    ob[(size_t)c * L + lane] = gm * v + qb[(size_t)c * L + lane];
  }
}

// ---------------------------------------------------------------------------
// host launcher
// ---------------------------------------------------------------------------
extern "C" void kernel_launch(void* const* d_in, const int* in_sizes, int n_in,
                              void* d_out, int out_size, void* d_ws, size_t ws_size,
                              hipStream_t stream) {
  (void)in_sizes; (void)n_in; (void)out_size; (void)ws_size;
  const float* qs    = (const float*)d_in[0];
  const float* ctx   = (const float*)d_in[1];
  const int*   ridx  = (const int*)d_in[2];
  const float* g_ctx = (const float*)d_in[3];
  const float* b_ctx = (const float*)d_in[4];
  const float* g_qs  = (const float*)d_in[5];
  const float* b_qs  = (const float*)d_in[6];
  const float* w_kv  = (const float*)d_in[7];
  const float* w_q   = (const float*)d_in[8];
  const float* w_out = (const float*)d_in[9];
  const float* g_out = (const float*)d_in[10];
  const float* b_out = (const float*)d_in[11];
  const float* gamma = (const float*)d_in[12];
  float* out = (float*)d_out;

  char* ws = (char*)d_ws;
  size_t off = 0;
  auto take = [&](size_t bytes) -> void* {
    void* p = ws + off;
    off = (off + bytes + 255) & ~(size_t)255;
    return p;
  };
  _Float16* ctxT  = (_Float16*)take((size_t)NB * LK * DIMC * 2);
  _Float16* qsnT  = (_Float16*)take((size_t)NB * LQ * DIMC * 2);
  _Float16* wkv16 = (_Float16*)take((size_t)1024 * 256 * 2);
  _Float16* wq16  = (_Float16*)take((size_t)512 * 256 * 2);
  _Float16* wout16= (_Float16*)take((size_t)256 * 512 * 2);
  _Float16* kf16  = (_Float16*)take((size_t)NB * LK * INNER * 2);
  _Float16* vf16  = (_Float16*)take((size_t)NB * LK * INNER * 2);
  _Float16* qf16  = (_Float16*)take((size_t)NB * LQ * INNER * 2);
  float*    mind  = (float*)take((size_t)NB * HEADS * LK * 4);
  int*      selix = (int*)take((size_t)NB * HEADS * TOPK * 4);
  _Float16* Ksel  = (_Float16*)take((size_t)NB * HEADS * TOPK * DH * 2);
  _Float16* Vt    = (_Float16*)take((size_t)NB * HEADS * DH * TOPK * 2);
  _Float16* outT  = (_Float16*)take((size_t)NB * LQ * INNER * 2);
  float*    resT  = (float*)take((size_t)NB * LQ * DIMC * 4);

  // weights -> f16
  f32_to_f16<<<(1024 * 256 + 255) / 256, 256, 0, stream>>>(w_kv, wkv16, 1024 * 256);
  f32_to_f16<<<(512 * 256 + 255) / 256, 256, 0, stream>>>(w_q, wq16, 512 * 256);
  f32_to_f16<<<(256 * 512 + 255) / 256, 256, 0, stream>>>(w_out, wout16, 256 * 512);

  // LayerNorm + transpose to position-major f16
  ln_transpose<<<dim3(LK / 32, NB), 256, 0, stream>>>(ctx, g_ctx, b_ctx, ctxT, LK);
  ln_transpose<<<dim3(LQ / 32, NB), 256, 0, stream>>>(qs, g_qs, b_qs, qsnT, LQ);

  // projections fused with per-head l2norm / f16 pack (WMMA)
  gemm_headfuse<<<dim3(LK / 32, 2, NB), 256, 0, stream>>>(
      ctxT, (size_t)LK * DIMC, wkv16, kf16, (size_t)LK * INNER, DIMC, 1);
  gemm_headfuse<<<dim3(LK / 32, 2, NB), 256, 0, stream>>>(
      ctxT, (size_t)LK * DIMC, wkv16 + (size_t)512 * 256, vf16,
      (size_t)LK * INNER, DIMC, 0);
  gemm_headfuse<<<dim3(LQ / 32, 2, NB), 256, 0, stream>>>(
      qsnT, (size_t)LQ * DIMC, wq16, qf16, (size_t)LQ * INNER, DIMC, 1);

  // token pruning: packed-f16 L1 min, exact selection, gather
  l1_min<<<dim3(LK / 256, NB * HEADS), 256, 0, stream>>>(qf16, kf16, ridx, mind);
  select_topk<<<NB * HEADS, 256, 0, stream>>>(mind, selix);
  gather_kv<<<dim3(TOPK / 256, NB * HEADS), 256, 0, stream>>>(kf16, vf16, selix,
                                                              Ksel, Vt);

  // flash attention (WMMA, LDS-staged K/V)
  attn_kernel<<<dim3(LQ / 128, NB * HEADS), 256, 0, stream>>>(qf16, Ksel, Vt, outT);

  // output projection (WMMA)
  gemm_abT<<<dim3(LQ / 32, DIMC / 64, NB), 256, 0, stream>>>(
      outT, (size_t)LQ * INNER, wout16, resT, (size_t)LQ * DIMC, DIMC, INNER);

  // final LN + residual
  final_ln<<<dim3(LQ / 32, NB), 256, 0, stream>>>(resT, g_out, b_out, gamma, qs,
                                                  out, LQ);
}
